// SeViMatch_33328946217687
// MI455X (gfx1250) — matrix operand order repaired and checked by reading, not compiled, not run on hardware
//
#include <hip/hip_runtime.h>
#include <math.h>

#define TOPK 4096
#define THRS 0.05f
#define EPSV 1e-12f

// ---------------- workspace layout (bytes) ----------------
#define OFFW_KH    0ull               // 4 inst * 1024*1024 f32 = 16 MB
#define OFFW_SC    16777216ull        // 4 inst * 1024*1024 f32 = 16 MB
#define OFFW_INV   33554432ull        // 4 inst * 128*128 f32
#define OFFW_SELK  33816576ull        // 4*4096 u32
#define OFFW_SELI  33882112ull        // 4*4096 u32
#define OFFW_FH    33947648ull        // 4*4096*64 f16
#define OFFW_M12I  36044800ull        // 2*4096 i32
#define OFFW_M21I  36077568ull        // 2*4096 i32
#define OFFW_ST    36110336ull        // 4 * 264 u32 radix-select state
#define ST_STRIDE  264                // [0]prefix [1]krem [2]G [3]cnt_gt [4]cnt_eq [8..263]hist

// ---------------- output layout (float elements) ----------------
#define OFF_COS  0ull
#define OFF_MK1  33554432ull
#define OFF_SC1  33570816ull
#define OFF_F1   33579008ull
#define OFF_MK2  34103296ull
#define OFF_SC2  34119680ull
#define OFF_F2   34127872ull
#define OFF_M12  34652160ull
#define OFF_MUT  34660352ull

typedef _Float16 v16h __attribute__((ext_vector_type(16)));
typedef _Float16 v8h  __attribute__((ext_vector_type(8)));
typedef float    v8f  __attribute__((ext_vector_type(8)));

__device__ __forceinline__ unsigned flipf(float f) {
  unsigned u = __float_as_uint(f);
  return (u & 0x80000000u) ? ~u : (u | 0x80000000u);
}
__device__ __forceinline__ float unflipf(unsigned k) {
  unsigned u = (k & 0x80000000u) ? (k ^ 0x80000000u) : ~k;
  return __uint_as_float(u);
}
__device__ __forceinline__ float cubw(float t) {
  const float a = -0.75f;
  t = fabsf(t);
  float w1 = ((a + 2.f) * t - (a + 3.f)) * t * t + 1.f;
  float w2 = ((a * t - 5.f * a) * t + 8.f * a) * t - 4.f * a;
  return t <= 1.f ? w1 : (t < 2.f ? w2 : 0.f);
}

// ---- init radix-select state ----
__global__ void k_init(unsigned* st) {
  int i = blockIdx.x * blockDim.x + threadIdx.x;
  if (i < 4 * ST_STRIDE) st[i] = ((i % ST_STRIDE) == 1) ? (unsigned)TOPK : 0u;
}

// ---- softmax(65) + 8x8 pixel shuffle -> kh[inst][1024][1024] ----
__global__ void k_softmax(const float* __restrict__ K1, const float* __restrict__ K2,
                          float* __restrict__ kh) {
  int gid = blockIdx.x * blockDim.x + threadIdx.x;   // 65536
  int inst = gid >> 14, pos = gid & 16383;
  int hc = pos >> 7, wc = pos & 127;
  int img = inst >> 1, b = inst & 1;
  const float* Kp = (img ? K2 : K1) + (size_t)b * 65 * 16384;
  float v[65];
  float mx = -INFINITY;
  for (int c = 0; c < 65; ++c) {
    v[c] = Kp[(size_t)c * 16384 + hc * 128 + wc];
    mx = fmaxf(mx, v[c]);
  }
  float s = 0.f;
  for (int c = 0; c < 65; ++c) { v[c] = expf(v[c] - mx); s += v[c]; }
  float inv = 1.f / s;
  float* khp = kh + (size_t)inst * 1048576;
  for (int c = 0; c < 64; ++c) {
    int i = c >> 3, j = c & 7;
    khp[(size_t)(hc * 8 + i) * 1024 + (wc * 8 + j)] = v[c] * inv;
  }
}

__device__ __forceinline__ float tap128(const float* p, int y, int x) {
  if (x < 0 || x >= 128 || y < 0 || y >= 128) return 0.f;
  return p[y * 128 + x];
}

// ---- 5x5 NMS + score = nearest(kh) * bilinear(Hrel); masked -> -1 ----
__global__ void k_nms_score(const float* __restrict__ kh,
                            const float* __restrict__ H1, const float* __restrict__ H2,
                            float* __restrict__ score) {
  int gid = blockIdx.x * blockDim.x + threadIdx.x;   // 4 * 1048576
  int inst = gid >> 20, p = gid & 1048575;
  int y = p >> 10, x = p & 1023;
  int img = inst >> 1, b = inst & 1;
  const float* khp = kh + (size_t)inst * 1048576;
  float c0 = khp[p];
  float lmax = -INFINITY;
  for (int dy = -2; dy <= 2; ++dy) {
    int yy = y + dy; if (yy < 0 || yy > 1023) continue;
    for (int dx = -2; dx <= 2; ++dx) {
      int xx = x + dx; if (xx < 0 || xx > 1023) continue;
      lmax = fmaxf(lmax, khp[yy * 1024 + xx]);
    }
  }
  bool mask = (c0 == lmax) && (c0 > THRS);
  float gx = 2.f * (float)x / 1023.f - 1.f;
  float gy = 2.f * (float)y / 1023.f - 1.f;
  // nearest sample of kh (Hf=Wf=1024)
  float ixn = ((gx + 1.f) * 1024.f - 1.f) * 0.5f;
  float iyn = ((gy + 1.f) * 1024.f - 1.f) * 0.5f;
  int xn = (int)rintf(ixn), yn = (int)rintf(iyn);
  float vk = 0.f;
  if (xn >= 0 && xn < 1024 && yn >= 0 && yn < 1024) vk = khp[yn * 1024 + xn];
  // bilinear sample of Hrel (128x128)
  const float* Hp = (img ? H2 : H1) + (size_t)b * 16384;
  float ixb = ((gx + 1.f) * 128.f - 1.f) * 0.5f;
  float iyb = ((gy + 1.f) * 128.f - 1.f) * 0.5f;
  float x0f = floorf(ixb), y0f = floorf(iyb);
  float wx = ixb - x0f, wy = iyb - y0f;
  int x0 = (int)x0f, y0 = (int)y0f;
  float vh = tap128(Hp, y0, x0) * (1.f - wx) * (1.f - wy)
           + tap128(Hp, y0, x0 + 1) * wx * (1.f - wy)
           + tap128(Hp, y0 + 1, x0) * (1.f - wx) * wy
           + tap128(Hp, y0 + 1, x0 + 1) * wx * wy;
  score[(size_t)inst * 1048576 + p] = mask ? (vk * vh) : -1.f;
}

// ---- radix-select pass: histogram of byte `pass` among prefix-matching keys ----
__global__ void k_hist(const float* __restrict__ score, unsigned* __restrict__ state, int pass) {
  int gid = blockIdx.x * blockDim.x + threadIdx.x;
  int inst = gid >> 20, e = gid & 1048575;
  unsigned* st = state + inst * ST_STRIDE;
  unsigned key = flipf(score[(size_t)inst * 1048576 + e]);
  if (pass > 0) {
    int hs = 32 - 8 * pass;
    if ((key >> hs) != (st[0] >> hs)) return;
  }
  unsigned by = (key >> (24 - 8 * pass)) & 0xFFu;
  atomicAdd(&st[8 + by], 1u);
}

__global__ void k_scan(unsigned* state, int pass) {
  unsigned* st = state + blockIdx.x * ST_STRIDE;
  if (threadIdx.x == 0) {
    unsigned need = st[1], cum = 0;
    for (int b2 = 255; b2 >= 0; --b2) {
      unsigned h = st[8 + b2];
      if (cum + h >= need) {
        st[2] += cum;                                  // G: strictly-greater accepted
        st[1] = need - cum;                            // remaining (ties at new prefix)
        st[0] |= ((unsigned)b2) << (24 - 8 * pass);
        break;
      }
      cum += h;
    }
  }
  __syncthreads();
  for (int i = threadIdx.x; i < 256; i += blockDim.x) st[8 + i] = 0;
}

// ---- compact: keys > T first, then exactly krem ties ----
__global__ void k_compact(const float* __restrict__ score, unsigned* __restrict__ state,
                          unsigned* __restrict__ selk, unsigned* __restrict__ seli) {
  int gid = blockIdx.x * blockDim.x + threadIdx.x;
  int inst = gid >> 20, e = gid & 1048575;
  unsigned* st = state + inst * ST_STRIDE;
  unsigned T = st[0];
  unsigned key = flipf(score[(size_t)inst * 1048576 + e]);
  if (key > T) {
    unsigned p = atomicAdd(&st[3], 1u);
    if (p < TOPK) { selk[inst * TOPK + p] = key; seli[inst * TOPK + p] = (unsigned)e; }
  } else if (key == T) {
    unsigned q = atomicAdd(&st[4], 1u);
    unsigned G = st[2];
    if (q < st[1]) { selk[inst * TOPK + G + q] = key; seli[inst * TOPK + G + q] = (unsigned)e; }
  }
}

// ---- bitonic sort 4096 (key desc, idx asc) per instance ----
__global__ void k_sort(unsigned* __restrict__ selk, unsigned* __restrict__ seli) {
  __shared__ unsigned sk[TOPK];
  __shared__ unsigned si[TOPK];
  int inst = blockIdx.x;
  for (int i = threadIdx.x; i < TOPK; i += blockDim.x) {
    sk[i] = selk[inst * TOPK + i];
    si[i] = seli[inst * TOPK + i];
  }
  __syncthreads();
  for (int k = 2; k <= TOPK; k <<= 1) {
    for (int j = k >> 1; j > 0; j >>= 1) {
      for (int i = threadIdx.x; i < TOPK; i += blockDim.x) {
        int ixj = i ^ j;
        if (ixj > i) {
          unsigned ka = sk[i], kb = sk[ixj], ia = si[i], ib = si[ixj];
          bool aFirst = (ka > kb) || (ka == kb && ia < ib);
          bool doSwap = ((i & k) == 0) ? !aFirst : aFirst;
          if (doSwap) { sk[i] = kb; sk[ixj] = ka; si[i] = ib; si[ixj] = ia; }
        }
      }
      __syncthreads();
    }
  }
  for (int i = threadIdx.x; i < TOPK; i += blockDim.x) {
    selk[inst * TOPK + i] = sk[i];
    seli[inst * TOPK + i] = si[i];
  }
}

// ---- write mk (x,y) and sc outputs ----
__global__ void k_mksc(const unsigned* __restrict__ selk, const unsigned* __restrict__ seli,
                       float* __restrict__ out) {
  int gid = blockIdx.x * blockDim.x + threadIdx.x;   // 16384
  int inst = gid >> 12, r = gid & 4095;
  int img = inst >> 1, b = inst & 1;
  unsigned idx = seli[inst * TOPK + r];
  unsigned key = selk[inst * TOPK + r];
  size_t mko = (img ? OFF_MK2 : OFF_MK1) + ((size_t)b * TOPK + r) * 2;
  size_t sco = (img ? OFF_SC2 : OFF_SC1) + (size_t)b * TOPK + r;
  out[mko + 0] = (float)(idx & 1023u);
  out[mko + 1] = (float)(idx >> 10);
  out[sco] = unflipf(key);
}

// ---- inverse channel norm of M at each (hc,wc) ----
__global__ void k_invnorm(const float* __restrict__ M1, const float* __restrict__ M2,
                          float* __restrict__ invn) {
  int gid = blockIdx.x * blockDim.x + threadIdx.x;   // 65536
  int inst = gid >> 14, pos = gid & 16383;
  int img = inst >> 1, b = inst & 1;
  const float* Mp = (img ? M2 : M1) + (size_t)b * 64 * 16384;
  float s = 0.f;
  for (int c = 0; c < 64; ++c) { float v = Mp[(size_t)c * 16384 + pos]; s += v * v; }
  invn[(size_t)inst * 16384 + pos] = 1.f / fmaxf(sqrtf(s), EPSV);
}

// ---- bicubic 64-d descriptor, normalize, emit f32 (out) + f16 (for WMMA) ----
__global__ void k_desc(const float* __restrict__ M1, const float* __restrict__ M2,
                       const float* __restrict__ invn, const unsigned* __restrict__ seli,
                       float* __restrict__ out, _Float16* __restrict__ fh) {
  int inst = blockIdx.x >> 12, kp = blockIdx.x & 4095;  // grid 16384, block 64
  int c = threadIdx.x;
  int img = inst >> 1, b = inst & 1;
  unsigned idx = seli[inst * TOPK + kp];
  float x = (float)(idx & 1023u), y = (float)(idx >> 10);
  float gx = 2.f * x / 1023.f - 1.f, gy = 2.f * y / 1023.f - 1.f;
  float ix = ((gx + 1.f) * 128.f - 1.f) * 0.5f;
  float iy = ((gy + 1.f) * 128.f - 1.f) * 0.5f;
  float x0f = floorf(ix), y0f = floorf(iy);
  float fx = ix - x0f, fy = iy - y0f;
  int x0 = (int)x0f, y0 = (int)y0f;
  const float* Mp = (img ? M2 : M1) + ((size_t)b * 64 + c) * 16384;
  const float* np = invn + (size_t)inst * 16384;
  float acc = 0.f;
  for (int dy = -1; dy <= 2; ++dy) {
    int yy = y0 + dy;
    float row = 0.f;
    for (int dx = -1; dx <= 2; ++dx) {
      int xx = x0 + dx;
      float v = 0.f;
      if (xx >= 0 && xx < 128 && yy >= 0 && yy < 128)
        v = Mp[yy * 128 + xx] * np[yy * 128 + xx];
      row += v * cubw(fx - (float)dx);
    }
    acc += row * cubw(fy - (float)dy);
  }
  __shared__ float red[64];
  red[c] = acc * acc;
  __syncthreads();
  for (int s = 32; s > 0; s >>= 1) { if (c < s) red[c] += red[c + s]; __syncthreads(); }
  float val = acc / fmaxf(sqrtf(red[0]), EPSV);
  size_t fo = (img ? OFF_F2 : OFF_F1) + ((size_t)b * TOPK + kp) * 64 + c;
  out[fo] = val;
  fh[((size_t)inst * TOPK + kp) * 64 + c] = (_Float16)val;
}

// ---- cossim GEMM via V_WMMA_F32_16X16X32_F16: one 16x16 tile per wave ----
__global__ void k_gemm(const _Float16* __restrict__ fh, float* __restrict__ cos_) {
  int lane = threadIdx.x & 31;
  int wave = threadIdx.x >> 5;
  int tile = blockIdx.x * 8 + wave;        // 131072 tiles = 2 * 256 * 256
  int b = tile >> 16;
  int tk = (tile >> 8) & 255;
  int tl = tile & 255;
  int m = lane & 15;
  int hi = lane >> 4;                       // half-wave select
  const _Float16* f1 = fh + (size_t)b * TOPK * 64;          // image1: inst = b
  const _Float16* f2 = fh + (size_t)(2 + b) * TOPK * 64;    // image2: inst = 2+b
  const _Float16* pa = f1 + (size_t)(tk * 16 + m) * 64;
  const _Float16* pb = f2 + (size_t)(tl * 16 + m) * 64;
  v8f c = {};
  #pragma unroll
  for (int kc = 0; kc < 64; kc += 32) {
    // A 16x32: lanes 0-15 hold K kc..kc+15; lanes 16-31 hold K kc+16..kc+31
    v16h a = *(const v16h*)(pa + kc + hi * 16);
    // B 32x16: lanes 0-15 hold K kc..kc+15; lanes 16-31 hold K kc+16..kc+31
    v16h bb = *(const v16h*)(pb + kc + hi * 16);
    c = __builtin_amdgcn_wmma_f32_16x16x32_f16(false, a, false, bb,
                                               (short)0, c, false, false);
  }
  int col = tl * 16 + m;
  int rbase = tk * 16 + (hi ? 8 : 0);
  #pragma unroll
  for (int r = 0; r < 8; ++r)
    cos_[((size_t)b * 4096 + rbase + r) * 4096 + col] = c[r];
}

// ---- argmax over rows (m12) ----
__global__ void k_m12(const float* __restrict__ cos_, int* __restrict__ m12i) {
  int b = blockIdx.x >> 12, k = blockIdx.x & 4095;
  const float* row = cos_ + ((size_t)b * 4096 + k) * 4096;
  float best = -INFINITY; int bi = 0x7fffffff;
  for (int l = threadIdx.x; l < 4096; l += blockDim.x) {
    float v = row[l];
    if (v > best || (v == best && l < bi)) { best = v; bi = l; }
  }
  __shared__ float bv[256];
  __shared__ int bx[256];
  bv[threadIdx.x] = best; bx[threadIdx.x] = bi;
  __syncthreads();
  for (int s = 128; s > 0; s >>= 1) {
    if (threadIdx.x < (unsigned)s) {
      float v2 = bv[threadIdx.x + s]; int i2 = bx[threadIdx.x + s];
      if (v2 > bv[threadIdx.x] || (v2 == bv[threadIdx.x] && i2 < bx[threadIdx.x])) {
        bv[threadIdx.x] = v2; bx[threadIdx.x] = i2;
      }
    }
    __syncthreads();
  }
  if (threadIdx.x == 0) m12i[b * 4096 + k] = bx[0];
}

// ---- argmax over columns (m21): thread per column, coalesced across lanes ----
__global__ void k_m21(const float* __restrict__ cos_, int* __restrict__ m21i) {
  int gid = blockIdx.x * blockDim.x + threadIdx.x;   // 8192
  int b = gid >> 12, l = gid & 4095;
  float best = -INFINITY; int bi = 0;
  for (int k = 0; k < 4096; ++k) {
    float v = cos_[((size_t)b * 4096 + k) * 4096 + l];
    if (v > best) { best = v; bi = k; }
  }
  m21i[gid] = bi;
}

__global__ void k_mutual(const int* __restrict__ m12i, const int* __restrict__ m21i,
                         float* __restrict__ out) {
  int gid = blockIdx.x * blockDim.x + threadIdx.x;   // 8192
  int b = gid >> 12, k = gid & 4095;
  int m = m12i[gid];
  out[OFF_M12 + gid] = (float)m;
  out[OFF_MUT + gid] = (m21i[b * 4096 + m] == k) ? 1.f : 0.f;
}

extern "C" void kernel_launch(void* const* d_in, const int* in_sizes, int n_in,
                              void* d_out, int out_size, void* d_ws, size_t ws_size,
                              hipStream_t stream) {
  (void)in_sizes; (void)n_in; (void)out_size; (void)ws_size;
  const float* M1 = (const float*)d_in[0];
  const float* K1 = (const float*)d_in[1];
  const float* H1 = (const float*)d_in[2];
  const float* M2 = (const float*)d_in[3];
  const float* K2 = (const float*)d_in[4];
  const float* H2 = (const float*)d_in[5];
  float* out = (float*)d_out;
  char* ws = (char*)d_ws;
  float*     kh    = (float*)(ws + OFFW_KH);
  float*     sc    = (float*)(ws + OFFW_SC);
  float*     invn  = (float*)(ws + OFFW_INV);
  unsigned*  selk  = (unsigned*)(ws + OFFW_SELK);
  unsigned*  seli  = (unsigned*)(ws + OFFW_SELI);
  _Float16*  fh    = (_Float16*)(ws + OFFW_FH);
  int*       m12i  = (int*)(ws + OFFW_M12I);
  int*       m21i  = (int*)(ws + OFFW_M21I);
  unsigned*  state = (unsigned*)(ws + OFFW_ST);

  k_init<<<(4 * ST_STRIDE + 255) / 256, 256, 0, stream>>>(state);
  k_softmax<<<256, 256, 0, stream>>>(K1, K2, kh);
  k_nms_score<<<16384, 256, 0, stream>>>(kh, H1, H2, sc);
  for (int p = 0; p < 4; ++p) {
    k_hist<<<16384, 256, 0, stream>>>(sc, state, p);
    k_scan<<<4, 256, 0, stream>>>(state, p);
  }
  k_compact<<<16384, 256, 0, stream>>>(sc, state, selk, seli);
  k_sort<<<4, 1024, 0, stream>>>(selk, seli);
  k_mksc<<<64, 256, 0, stream>>>(selk, seli, out);
  k_invnorm<<<256, 256, 0, stream>>>(M1, M2, invn);
  k_desc<<<16384, 64, 0, stream>>>(M1, M2, invn, seli, out, fh);
  k_gemm<<<16384, 256, 0, stream>>>(fh, out + OFF_COS);
  k_m12<<<8192, 256, 0, stream>>>(out + OFF_COS, m12i);
  k_m21<<<32, 256, 0, stream>>>(out + OFF_COS, m21i);
  k_mutual<<<32, 256, 0, stream>>>(m12i, m21i, out);
}
